// AttentivePool_61168924229741
// MI455X (gfx1250) — matrix operand order repaired and checked by reading, not compiled, not run on hardware
//
#include <hip/hip_runtime.h>

typedef __attribute__((ext_vector_type(16))) _Float16 v16h;
typedef __attribute__((ext_vector_type(8)))  _Float16 v8h;
typedef __attribute__((ext_vector_type(8)))  float    v8f;
typedef __attribute__((ext_vector_type(4)))  float    v4f;

#define BB  16
#define SS  8192
#define HH  768
#define NH  12
#define NHP 16
#define HD  64
#define KSEG 8           // K-split factor for the pooling pass

// workspace layout (bytes)
#define WS_QK    0u                        // _Float16[NHP*HH]      24576 B
#define WS_QB    24576u                    // float[NHP]               64 B
#define WS_GATE  24704u                    // float[HH]              3072 B
#define WS_SCORE 32768u                    // float[BB*SS*NHP]    8388608 B
#define WS_ATTN  (32768u + 8388608u)       // _Float16[BB*NHP*SS] 4194304 B
#define WS_POOL  (32768u + 8388608u + 4194304u)      // float[BB*NHP*HH]  786432 B
#define WS_PART  (WS_POOL + 786432u)       // float[BB*KSEG*NHP*HH] 6291456 B

// ---------------------------------------------------------------------------
// Kernel 1: prep.  qk[h][i] = scale * sum_d q[h,d]*w_kv[h*64+d, i]  (f16)
//                  qb[h]    = scale * sum_d q[h,d]*b_kv[h*64+d]
//                  gate[m]  = sigmoid(q . w_gate[m] + b_gate[m])
// Pad heads 12..15 are zeroed so WMMA pad columns/rows are inert.
// ---------------------------------------------------------------------------
__global__ void __launch_bounds__(256)
prep_kernel(const float* __restrict__ query, const float* __restrict__ w_kv,
            const float* __restrict__ b_kv, const float* __restrict__ w_gate,
            const float* __restrict__ b_gate, _Float16* __restrict__ qk,
            float* __restrict__ qb, float* __restrict__ gate) {
  const float scale = 0.125f; // 1/sqrt(64)
  const int tot = NHP * HH + NHP + HH;
  for (int t = blockIdx.x * blockDim.x + threadIdx.x; t < tot;
       t += gridDim.x * blockDim.x) {
    if (t < NHP * HH) {
      int h = t / HH, i = t - h * HH;
      float s = 0.f;
      if (h < NH) {
        for (int d = 0; d < HD; ++d)
          s += query[h * HD + d] * w_kv[(size_t)(h * HD + d) * HH + i];
        s *= scale;
      }
      qk[h * HH + i] = (_Float16)s;
    } else if (t < NHP * HH + NHP) {
      int h = t - NHP * HH;
      float s = 0.f;
      if (h < NH)
        for (int d = 0; d < HD; ++d) s += query[h * HD + d] * b_kv[h * HD + d];
      qb[h] = s * scale;
    } else {
      int m = t - NHP * HH - NHP;
      float s = b_gate[m];
      for (int i = 0; i < HH; ++i) s += query[i] * w_gate[(size_t)m * HH + i];
      gate[m] = 1.f / (1.f + __expf(-s));
    }
  }
}

// ---------------------------------------------------------------------------
// Kernel 2: scores via WMMA.  One wave -> 16 s-rows x 16 heads, K=768.
// A: 16 rows of x (f16-converted).  B: qk^T (head = N column).
// scores stored row-major (B, S, NHP) f32.
// ---------------------------------------------------------------------------
__global__ void __launch_bounds__(256)
score_kernel(const float* __restrict__ x, const _Float16* __restrict__ qk,
             const float* __restrict__ qb, float* __restrict__ scores) {
  int tid = threadIdx.x;
  int wave = tid >> 5, lane = tid & 31;
  int tile = blockIdx.x * 8 + wave;  // 0..8191  (B*S/16 tiles)
  int b = tile >> 9;                 // / (SS/16)
  int srow = (tile & 511) << 4;
  int m = lane & 15, sel = lane >> 4, n = m;

  const float* xrow = x + ((size_t)b * SS + srow + m) * HH;
  const _Float16* bcol = qk + n * HH;  // B[k][n] = qk[n][k]

  v8f acc = {};
#pragma unroll 4
  for (int kc = 0; kc < 24; ++kc) {
    int k0 = kc * 32;
    // A fragment: lanes 0-15 hold K 0-15, lanes 16-31 hold K 16-31
    // (contiguous, matching B), for row M = lane&15.
    v4f f0 = *(const v4f*)(xrow + k0 + sel * 16);
    v4f f1 = *(const v4f*)(xrow + k0 + sel * 16 + 4);
    v4f f2 = *(const v4f*)(xrow + k0 + sel * 16 + 8);
    v4f f3 = *(const v4f*)(xrow + k0 + sel * 16 + 12);
    v16h a;
#pragma unroll
    for (int j = 0; j < 4; ++j) {
      a[j]      = (_Float16)f0[j];
      a[4 + j]  = (_Float16)f1[j];
      a[8 + j]  = (_Float16)f2[j];
      a[12 + j] = (_Float16)f3[j];
    }
    // B fragment: lane holds contiguous K run [sel*16, sel*16+16) of col n.
    v16h bf = *(const v16h*)(bcol + k0 + sel * 16);
    acc = __builtin_amdgcn_wmma_f32_16x16x32_f16(false, a, false, bf, (short)0,
                                                 acc, false, false);
  }
  float qbn = qb[n];
  size_t base = ((size_t)b * SS + srow + sel * 8) * NHP + n;
#pragma unroll
  for (int r = 0; r < 8; ++r)
    scores[base + (size_t)r * NHP] = acc[r] + qbn;
}

// ---------------------------------------------------------------------------
// Kernel 3: softmax over S per (b,h).  Writes f16 attn (B, NHP, S);
// pad heads h>=12 are written as zeros.
// ---------------------------------------------------------------------------
__global__ void __launch_bounds__(256)
softmax_kernel(const float* __restrict__ scores, _Float16* __restrict__ attn) {
  __shared__ float ev[SS];
  __shared__ float red[256];
  int tid = threadIdx.x;
  int b = blockIdx.x >> 4;
  int h = blockIdx.x & 15;
  _Float16* arow = attn + ((size_t)b * NHP + h) * SS;
  if (h >= NH) {
    for (int s = tid; s < SS; s += 256) arow[s] = (_Float16)0.f;
    return;
  }
  const float* srcb = scores + (size_t)b * SS * NHP + h;
  float lmax = -1e30f;
  for (int s = tid; s < SS; s += 256) {
    float v = srcb[(size_t)s * NHP];
    ev[s] = v;
    lmax = fmaxf(lmax, v);
  }
  red[tid] = lmax;
  __syncthreads();
  for (int off = 128; off > 0; off >>= 1) {
    if (tid < off) red[tid] = fmaxf(red[tid], red[tid + off]);
    __syncthreads();
  }
  float gmax = red[0];
  __syncthreads();
  float lsum = 0.f;
  for (int s = tid; s < SS; s += 256) {
    float e = __expf(ev[s] - gmax);
    ev[s] = e;
    lsum += e;
  }
  red[tid] = lsum;
  __syncthreads();
  for (int off = 128; off > 0; off >>= 1) {
    if (tid < off) red[tid] += red[tid + off];
    __syncthreads();
  }
  float inv = 1.f / red[0];
  for (int s = tid; s < SS; s += 256) arow[s] = (_Float16)(ev[s] * inv);
}

// ---------------------------------------------------------------------------
// Kernel 4: partial pooled = attn_b(16 x Kseg) @ x_b(Kseg x 768) via WMMA.
// Block = (b, 128-col group, K segment); 8 waves -> 8 N-tiles of 16 cols.
// x staged transposed into double-buffered LDS as f16 (one barrier/chunk);
// global prefetch for chunk i+1 is issued before the WMMA of chunk i.
// ---------------------------------------------------------------------------
__global__ void __launch_bounds__(256)
pool_kernel(const float* __restrict__ x, const _Float16* __restrict__ attn,
            float* __restrict__ part) {
  __shared__ _Float16 xt[2][128 * 32];  // [buf][col][k], 2 x 8 KB
  int tid = threadIdx.x;
  int wave = tid >> 5, lane = tid & 31;
  int blk = blockIdx.x;            // BB * 6 * KSEG = 768 blocks
  int b = blk / (6 * KSEG);
  int rem = blk % (6 * KSEG);
  int cgrp = rem >> 3;             // 0..5
  int seg = rem & (KSEG - 1);      // 0..7
  int c0 = cgrp * 128;
  int m = lane & 15, sel = lane >> 4, n = m;

  int krow = tid >> 3;             // 0..31 : k within chunk
  int cg = (tid & 7) * 16;         // 16-column group this thread transposes
  const _Float16* arow = attn + ((size_t)b * NHP + m) * SS;
  const size_t xbase = (size_t)b * SS * HH + c0 + cg;
  const int kc0 = seg * 32;        // 32 chunks of K=32 per segment

  // prologue: stage chunk kc0 into buffer 0
  {
    const float* src = x + xbase + (size_t)(kc0 * 32 + krow) * HH;
    v4f f0 = *(const v4f*)(src);
    v4f f1 = *(const v4f*)(src + 4);
    v4f f2 = *(const v4f*)(src + 8);
    v4f f3 = *(const v4f*)(src + 12);
    _Float16* xw = &xt[0][cg * 32 + krow];
#pragma unroll
    for (int j = 0; j < 4; ++j) {
      xw[j * 32]        = (_Float16)f0[j];
      xw[(4 + j) * 32]  = (_Float16)f1[j];
      xw[(8 + j) * 32]  = (_Float16)f2[j];
      xw[(12 + j) * 32] = (_Float16)f3[j];
    }
  }
  __syncthreads();

  v8f acc = {};
  int p = 0;
  for (int i = 0; i < 32; ++i) {
    int kc = kc0 + i;
    // prefetch next chunk from global (hidden under this chunk's WMMA)
    v4f f0, f1, f2, f3;
    if (i < 31) {
      const float* src = x + xbase + (size_t)((kc + 1) * 32 + krow) * HH;
      f0 = *(const v4f*)(src);
      f1 = *(const v4f*)(src + 4);
      f2 = *(const v4f*)(src + 8);
      f3 = *(const v4f*)(src + 12);
    }
    // A fragment from f16 attn (L2-resident, 256 KB per b)
    int k0 = kc * 32;
    v8h a0 = *(const v8h*)(arow + k0 + sel * 16);
    v8h a1 = *(const v8h*)(arow + k0 + sel * 16 + 8);
    v16h a;
#pragma unroll
    for (int j = 0; j < 8; ++j) {
      a[j] = a0[j];
      a[8 + j] = a1[j];
    }
    // B fragment: contiguous K-16 run of LDS column (wave*16 + n)
    v16h bf = *(const v16h*)(&xt[p][(wave * 16 + n) * 32 + sel * 16]);
    acc = __builtin_amdgcn_wmma_f32_16x16x32_f16(false, a, false, bf, (short)0,
                                                 acc, false, false);
    // fill the other buffer for the next iteration
    if (i < 31) {
      _Float16* xw = &xt[p ^ 1][cg * 32 + krow];
#pragma unroll
      for (int j = 0; j < 4; ++j) {
        xw[j * 32]        = (_Float16)f0[j];
        xw[(4 + j) * 32]  = (_Float16)f1[j];
        xw[(8 + j) * 32]  = (_Float16)f2[j];
        xw[(12 + j) * 32] = (_Float16)f3[j];
      }
      __syncthreads();
    }
    p ^= 1;
  }
  size_t base =
      (((size_t)b * KSEG + seg) * NHP + sel * 8) * HH + c0 + wave * 16 + n;
#pragma unroll
  for (int r = 0; r < 8; ++r)
    part[base + (size_t)r * HH] = acc[r];
}

// ---------------------------------------------------------------------------
// Kernel 4b: deterministic fixed-order reduction of the K-split partials.
// ---------------------------------------------------------------------------
__global__ void __launch_bounds__(256)
reduce_kernel(const float* __restrict__ part, float* __restrict__ pooled) {
  int idx = blockIdx.x * 256 + threadIdx.x;  // < BB*NHP*HH = 196608
  int b = idx / (NHP * HH);
  int r = idx - b * (NHP * HH);
  float s = 0.f;
#pragma unroll
  for (int seg = 0; seg < KSEG; ++seg)
    s += part[((size_t)b * KSEG + seg) * (NHP * HH) + r];
  pooled[(size_t)b * (NHP * HH) + r] = s;
}

// ---------------------------------------------------------------------------
// Kernel 5: o = (pooled @ w_v.T + b_v) concat-heads @ w_out.T + b_out, gated.
// One block per batch element; tiny.
// ---------------------------------------------------------------------------
__global__ void __launch_bounds__(256)
final_kernel(const float* __restrict__ pooled, const float* __restrict__ w_kv,
             const float* __restrict__ b_kv, const float* __restrict__ w_out,
             const float* __restrict__ b_out, const float* __restrict__ gate,
             float* __restrict__ out) {
  __shared__ float oc[HH];
  int tid = threadIdx.x;
  int b = blockIdx.x;
  for (int j = tid; j < HH; j += 256) {
    int h = j >> 6;
    const float* pr = pooled + ((size_t)b * NHP + h) * HH;
    const float* wr = w_kv + (size_t)(HH + j) * HH;  // value rows
    float s = b_kv[HH + j];
    for (int i = 0; i < HH; ++i) s += pr[i] * wr[i];
    oc[j] = s;
  }
  __syncthreads();
  for (int mo = tid; mo < HH; mo += 256) {
    const float* wo = w_out + (size_t)mo * HH;
    float s = b_out[mo];
    for (int i = 0; i < HH; ++i) s += oc[i] * wo[i];
    out[(size_t)b * HH + mo] = s * gate[mo];
  }
}

extern "C" void kernel_launch(void* const* d_in, const int* in_sizes, int n_in,
                              void* d_out, int out_size, void* d_ws,
                              size_t ws_size, hipStream_t stream) {
  const float* x      = (const float*)d_in[0];
  const float* query  = (const float*)d_in[1];
  const float* w_kv   = (const float*)d_in[2];
  const float* b_kv   = (const float*)d_in[3];
  const float* w_out  = (const float*)d_in[4];
  const float* b_out  = (const float*)d_in[5];
  const float* w_gate = (const float*)d_in[6];
  const float* b_gate = (const float*)d_in[7];

  char* ws = (char*)d_ws;
  _Float16* qk     = (_Float16*)(ws + WS_QK);
  float*    qb     = (float*)(ws + WS_QB);
  float*    gate   = (float*)(ws + WS_GATE);
  float*    scores = (float*)(ws + WS_SCORE);
  _Float16* attn   = (_Float16*)(ws + WS_ATTN);
  float*    pooled = (float*)(ws + WS_POOL);
  float*    part   = (float*)(ws + WS_PART);
  float*    out    = (float*)d_out;

  prep_kernel<<<52, 256, 0, stream>>>(query, w_kv, b_kv, w_gate, b_gate, qk,
                                      qb, gate);
  score_kernel<<<1024, 256, 0, stream>>>(x, qk, qb, scores);
  softmax_kernel<<<BB * NHP, 256, 0, stream>>>(scores, attn);
  pool_kernel<<<BB * 6 * KSEG, 256, 0, stream>>>(x, attn, part);
  reduce_kernel<<<768, 256, 0, stream>>>(part, pooled);
  final_kernel<<<BB, 256, 0, stream>>>(pooled, w_kv, b_kv, w_out, b_out, gate,
                                       out);
}
